// CapsuleNetwork_83769042141864
// MI455X (gfx1250) — matrix operand order repaired and checked by reading, not compiled, not run on hardware
//
#include <hip/hip_runtime.h>
#include <hip/hip_bf16.h>

#define BATCH 256
#define EPSF 1e-7f

typedef __bf16 bf16_t;
typedef bf16_t v16bf __attribute__((ext_vector_type(16)));
typedef float  v8f   __attribute__((ext_vector_type(8)));
typedef float  v2f   __attribute__((ext_vector_type(2)));
typedef int    v4i_  __attribute__((vector_size(4 * sizeof(int))));

union BF16Frag { v16bf v; unsigned int u[8]; };

static __device__ inline unsigned short f2bf(float f) {
  unsigned int u = __builtin_bit_cast(unsigned int, f);
  unsigned int r = u + 0x7FFFu + ((u >> 16) & 1u);   // round-to-nearest-even
  return (unsigned short)(r >> 16);
}

// ---- async global->LDS (ASYNCcnt-tracked) helpers ------------------------
static __device__ __forceinline__ void async_cp16(const void* src, void* lds_dst) {
#if __has_builtin(__builtin_amdgcn_global_load_async_to_lds_b128)
  __builtin_amdgcn_global_load_async_to_lds_b128(
      (__attribute__((address_space(1))) v4i_*)(unsigned long long)src,
      (__attribute__((address_space(3))) v4i_*)(unsigned int)(unsigned long long)lds_dst,
      0, 0);
#else
  unsigned int d = (unsigned int)(unsigned long long)lds_dst;   // LDS byte offset
  unsigned long long s = (unsigned long long)src;
  asm volatile("global_load_async_to_lds_b128 %0, %1, off"
               :: "v"(d), "v"(s) : "memory");
#endif
}

static __device__ __forceinline__ void wait_async5() {
#if __has_builtin(__builtin_amdgcn_s_wait_asynccnt)
  __builtin_amdgcn_s_wait_asynccnt(5);
#else
  asm volatile("s_wait_asynccnt 0x5" ::: "memory");
#endif
}
static __device__ __forceinline__ void wait_async0() {
#if __has_builtin(__builtin_amdgcn_s_wait_asynccnt)
  __builtin_amdgcn_s_wait_asynccnt(0);
#else
  asm volatile("s_wait_asynccnt 0x0" ::: "memory");
#endif
}

// ---------------------------------------------------------------------------
// conv1: X (B,28,28,1) * K1 (9,9,1,256) stride 1 -> relu -> bf16 (B,20,20,256)
// ---------------------------------------------------------------------------
__global__ void conv1_kernel(const float* __restrict__ X,
                             const float* __restrict__ K1,
                             const float* __restrict__ bias,
                             unsigned short* __restrict__ c1bf) {
  __shared__ float patch[81];
  int bp = blockIdx.x;                 // b*400 + pos
  int b = bp / 400, pos = bp % 400;
  int h = pos / 20, w = pos % 20;
  int t = threadIdx.x;
  if (t < 81) {
    int kh = t / 9, kw = t % 9;
    patch[t] = X[b * 784 + (h + kh) * 28 + (w + kw)];
  }
  __syncthreads();
  float s = bias[t];
  #pragma unroll
  for (int i = 0; i < 81; ++i) s += patch[i] * K1[i * 256 + t];
  s = fmaxf(s, 0.f);
  c1bf[b * 102400 + pos * 256 + t] = f2bf(s);
}

// ---------------------------------------------------------------------------
// pack conv2 weights (9,9,256,256) f32 -> bf16 K-pairs: wk2[(k/2)*256+oc]
// ---------------------------------------------------------------------------
__global__ void conv2w_kernel(const float* __restrict__ K2,
                              unsigned int* __restrict__ wk2) {
  int t = blockIdx.x * blockDim.x + threadIdx.x;   // 10368*256 exactly
  int p = t >> 8;
  int oc = t & 255;
  float w0 = K2[(p * 2 + 0) * 256 + oc];
  float w1 = K2[(p * 2 + 1) * 256 + oc];
  wk2[t] = (unsigned int)f2bf(w0) | ((unsigned int)f2bf(w1) << 16);
}

// ---------------------------------------------------------------------------
// conv2 implicit GEMM: M=9216, N=256, K=20736 ; bf16 WMMA 16x16x32
// Block = 256 thr (8 waves) computes one 16(M) x 256(N) strip, wave w owns
// N-tiles {w, w+8}. K-chunks of 32 double-buffered through LDS with async
// global->LDS b128 copies (exactly 5 async instructions per wave per chunk:
// 4 B-tile segments + 1 lane-masked A-tile segment) + s_wait_asynccnt.
// ---------------------------------------------------------------------------
#define BSTRIDE 260   // LDS dwords per B pair-row (pad: halves hit disjoint banks)
#define ASTRIDE 20    // LDS dwords per A row

static __device__ __forceinline__ void conv2_copy_chunk(
    const unsigned short* __restrict__ c1bf,
    const unsigned int* __restrict__ wk2,
    unsigned int* ldsB, unsigned int* ldsA,
    int kc, int mbase, int t, int wid, int lane) {
  // B tile: 16 pair-rows x 256 dwords, contiguous in wk2
  #pragma unroll
  for (int i = 0; i < 4; ++i) {
    int s = i * 256 + t;          // 0..1023 segments of 16B
    int r = s >> 6;
    int seg = s & 63;
    async_cp16(wk2 + (kc * 16 + r) * 256 + seg * 4,
               ldsB + r * BSTRIDE + seg * 4);
  }
  // A tile: 16 rows x 64B (im2col gather); 64 segments, 8 per wave
  if (lane < 8) {
    int slot = wid * 8 + lane;    // 0..63
    int r = slot >> 2, j = slot & 3;
    int m = mbase + r;
    int b = m / 36, pos = m % 36;
    int oh = pos / 6, ow = pos % 6;
    int khw = kc >> 3;
    int kh = khw / 9, kw = khw - kh * 9;
    int ic0 = (kc & 7) * 32;
    async_cp16(c1bf + b * 102400 +
                   ((oh * 2 + kh) * 20 + (ow * 2 + kw)) * 256 + ic0 + j * 8,
               ldsA + r * ASTRIDE + j * 4);
  }
}

__global__ void conv2_wmma_kernel(const unsigned short* __restrict__ c1bf,
                                  const unsigned int* __restrict__ wk2,
                                  const float* __restrict__ bias,
                                  float* __restrict__ c2) {
  __shared__ __align__(16) unsigned int ldsB[2][16 * BSTRIDE];
  __shared__ __align__(16) unsigned int ldsA[2][16 * ASTRIDE];

  const int t    = threadIdx.x;
  const int lane = t & 31;
  const int wid  = t >> 5;                     // 0..7
  const int half = (lane >> 4) & 1;
  const int lmod = lane & 15;
  const int mbase = blockIdx.x * 16;
  const int oc0 = wid * 16 + lmod;             // N-tile w
  const int oc1 = oc0 + 128;                   // N-tile w+8

  v8f acc0 = {}, acc1 = {};
  int buf = 0;
  conv2_copy_chunk(c1bf, wk2, ldsB[0], ldsA[0], 0, mbase, t, wid, lane);

  for (int kc = 0; kc < 648; ++kc) {
    if (kc + 1 < 648) {
      conv2_copy_chunk(c1bf, wk2, ldsB[buf ^ 1], ldsA[buf ^ 1],
                       kc + 1, mbase, t, wid, lane);
      wait_async5();                           // current chunk's 5 are done
    } else {
      wait_async0();
    }
    __syncthreads();                           // all waves' copies visible

    BF16Frag A, B0, B1;
    const unsigned int* la = ldsA[buf];
    const unsigned int* lb = ldsB[buf];
    #pragma unroll
    for (int v = 0; v < 8; ++v) {
      int pa = (v < 4) ? (half * 4 + v) : (8 + half * 4 + (v - 4));
      A.u[v]  = la[lmod * ASTRIDE + pa];
      int pb = (v < 4) ? (half * 4 + v) : (8 + half * 4 + (v - 4));
      B0.u[v] = lb[pb * BSTRIDE + oc0];
      B1.u[v] = lb[pb * BSTRIDE + oc1];
    }
    acc0 = __builtin_amdgcn_wmma_f32_16x16x32_bf16(false, A.v, false, B0.v,
                                                   (short)0, acc0, false, false);
    acc1 = __builtin_amdgcn_wmma_f32_16x16x32_bf16(false, A.v, false, B1.v,
                                                   (short)0, acc1, false, false);
    __syncthreads();                           // done reading buf
    buf ^= 1;
  }

  #pragma unroll
  for (int i = 0; i < 8; ++i) {
    int row = mbase + i + half * 8;
    int bb = row / 36, pp = row % 36;
    float v0 = fmaxf(acc0[i] + bias[oc0], 0.f);
    float v1 = fmaxf(acc1[i] + bias[oc1], 0.f);
    c2[bb * 9216 + pp * 256 + oc0] = v0;
    c2[bb * 9216 + pp * 256 + oc1] = v1;
  }
}

// ---------------------------------------------------------------------------
// squash caps1: (B,1152,8) rows; c2 flat layout == caps1 flat layout
// ---------------------------------------------------------------------------
__global__ void squash1_kernel(const float* __restrict__ c2,
                               float* __restrict__ caps1) {
  int t = blockIdx.x * blockDim.x + threadIdx.x;   // b*1152+n
  const float* s = c2 + (long)t * 8;
  float v[8], sq = 0.f;
  #pragma unroll
  for (int i = 0; i < 8; ++i) { v[i] = s[i]; sq += v[i] * v[i]; }
  float scale = sq / (1.f + sq) * rsqrtf(sq + EPSF);
  float* d = caps1 + (long)t * 8;
  #pragma unroll
  for (int i = 0; i < 8; ++i) d[i] = v[i] * scale;
}

// ---------------------------------------------------------------------------
// u_hat = einsum('njod,bnd->bnjo'): per n, GEMM (256 x 8) @ (8 x 160)
// ---------------------------------------------------------------------------
__global__ void uhat_wmma_kernel(const float* __restrict__ caps1,
                                 const float* __restrict__ W,
                                 float* __restrict__ uhat) {
  const int n = blockIdx.x;
  const int mbase = blockIdx.y * 16;
  const int j = threadIdx.x >> 5;                  // 0..9
  const int lane = threadIdx.x & 31;
  const int half = (lane >> 4) & 1;
  const int lmod = lane & 15;
  const float* Wn = W + ((long)n * 10 + j) * 128;
  v8f acc = {};
  #pragma unroll
  for (int kstep = 0; kstep < 8; kstep += 4) {
    v2f a, bv;
    int kb = kstep + 2 * half;
    int row = mbase + lmod;
    a.x = caps1[(long)row * 9216 + n * 8 + kb];
    a.y = caps1[(long)row * 9216 + n * 8 + kb + 1];
    bv.x = Wn[lmod * 8 + kb];
    bv.y = Wn[lmod * 8 + kb + 1];
    acc = __builtin_amdgcn_wmma_f32_16x16x4_f32(false, a, false, bv,
                                                (short)0, acc, false, false);
  }
  #pragma unroll
  for (int i = 0; i < 8; ++i) {
    int bb = mbase + i + half * 8;
    uhat[(((long)bb * 1152 + n) * 10 + j) * 16 + lmod] = acc[i];
  }
}

// ---------------------------------------------------------------------------
__global__ void v1_kernel(const float* __restrict__ uhat,
                          float* __restrict__ v1) {
  __shared__ float s1l[160];
  __shared__ float sqs[10];
  int b = blockIdx.x, t = threadIdx.x;
  float s = 0.f;
  const float* up = uhat + (long)b * 1152 * 160 + t;
  for (int n = 0; n < 1152; ++n) s += up[(long)n * 160];
  s *= 0.1f;
  s1l[t] = s;
  __syncthreads();
  if (t < 10) {
    float q = 0.f;
    for (int o = 0; o < 16; ++o) { float x = s1l[t * 16 + o]; q += x * x; }
    sqs[t] = q;
  }
  __syncthreads();
  float q = sqs[t >> 4];
  v1[b * 160 + t] = s * (q / (1.f + q) * rsqrtf(q + EPSF));
}

// ---------------------------------------------------------------------------
__global__ void routing_kernel(const float* __restrict__ uhat,
                               const float* __restrict__ v1,
                               float* __restrict__ s2) {
  __shared__ float v1l[160];
  __shared__ float s2l[160];
  int b = blockIdx.x, t = threadIdx.x;
  if (t < 160) { v1l[t] = v1[b * 160 + t]; s2l[t] = 0.f; }
  __syncthreads();
  for (int n = t; n < 1152; n += 256) {
    const float* up = uhat + ((long)b * 1152 + n) * 160;
    float lg[10], m = -3.0e38f;
    #pragma unroll
    for (int j = 0; j < 10; ++j) {
      float d = 0.f;
      for (int o = 0; o < 16; ++o) d += up[j * 16 + o] * v1l[j * 16 + o];
      lg[j] = d; m = fmaxf(m, d);
    }
    float sum = 0.f;
    #pragma unroll
    for (int j = 0; j < 10; ++j) { lg[j] = __expf(lg[j] - m); sum += lg[j]; }
    float inv = 1.f / sum;
    #pragma unroll
    for (int j = 0; j < 10; ++j) {
      float c = lg[j] * inv;
      for (int o = 0; o < 16; ++o)
        atomicAdd(&s2l[j * 16 + o], c * up[j * 16 + o]);
    }
  }
  __syncthreads();
  if (t < 160) s2[b * 160 + t] = s2l[t];
}

// ---------------------------------------------------------------------------
__global__ void v2_kernel(const float* __restrict__ s2,
                          const int* __restrict__ Y,
                          float* __restrict__ caps2_out,
                          float* __restrict__ ypred,
                          float* __restrict__ dec_in) {
  __shared__ float sqs[10];
  __shared__ float v2l[160];
  __shared__ float proba[10];
  int b = blockIdx.x, t = threadIdx.x;
  int j = t >> 4;
  float s = s2[b * 160 + t];
  if (t < 10) {
    float q = 0.f;
    for (int o = 0; o < 16; ++o) { float x = s2[b * 160 + t * 16 + o]; q += x * x; }
    sqs[t] = q;
  }
  __syncthreads();
  float q = sqs[j];
  float v = s * (q / (1.f + q) * rsqrtf(q + EPSF));
  v2l[t] = v;
  caps2_out[b * 160 + t] = v;
  __syncthreads();
  if (t < 10) {
    float q2 = 0.f;
    for (int o = 0; o < 16; ++o) { float x = v2l[t * 16 + o]; q2 += x * x; }
    proba[t] = sqrtf(q2 + EPSF);
  }
  __syncthreads();
  if (t == 0) {
    int best = 0; float bv = proba[0];
    for (int jj = 1; jj < 10; ++jj) if (proba[jj] > bv) { bv = proba[jj]; best = jj; }
    ypred[b] = (float)best;
  }
  dec_in[b * 160 + t] = (j == Y[b]) ? v : 0.f;
}

// ---------------------------------------------------------------------------
// dense layer: C = act(A(MxK) @ W(KxN) + bias) via f32 WMMA 16x16x4
// ---------------------------------------------------------------------------
__global__ void gemm_f32_wmma_kernel(const float* __restrict__ A,
                                     const float* __restrict__ Wt,
                                     const float* __restrict__ bias,
                                     float* __restrict__ C,
                                     int M, int K, int N, int act) {
  const int lane = threadIdx.x & 31;
  const int wid  = threadIdx.x >> 5;
  const int half = (lane >> 4) & 1;
  const int lmod = lane & 15;
  const int ntile = blockIdx.x * 4 + wid;
  if (ntile * 16 >= N) return;
  const int mbase = blockIdx.y * 16;
  const int col = ntile * 16 + lmod;
  const int row = mbase + lmod;
  v8f acc = {};
  for (int k = 0; k < K; k += 4) {
    v2f a, bv;
    int kb = k + 2 * half;
    a.x = A[(long)row * K + kb];
    a.y = A[(long)row * K + kb + 1];
    bv.x = Wt[(long)kb * N + col];
    bv.y = Wt[(long)(kb + 1) * N + col];
    acc = __builtin_amdgcn_wmma_f32_16x16x4_f32(false, a, false, bv,
                                                (short)0, acc, false, false);
  }
  #pragma unroll
  for (int i = 0; i < 8; ++i) {
    int r = mbase + i + half * 8;
    float val = acc[i] + bias[col];
    val = (act == 0) ? fmaxf(val, 0.f) : 1.f / (1.f + __expf(-val));
    C[(long)r * N + col] = val;
  }
}

// ---------------------------------------------------------------------------
extern "C" void kernel_launch(void* const* d_in, const int* in_sizes, int n_in,
                              void* d_out, int out_size, void* d_ws, size_t ws_size,
                              hipStream_t stream) {
  (void)in_sizes; (void)n_in; (void)out_size; (void)ws_size;
  const float* X   = (const float*)d_in[0];
  const int*   Y   = (const int*)d_in[1];
  const float* K1  = (const float*)d_in[2];
  const float* c1b = (const float*)d_in[3];
  const float* K2  = (const float*)d_in[4];
  const float* c2b = (const float*)d_in[5];
  const float* Wc  = (const float*)d_in[6];
  const float* W1  = (const float*)d_in[7];
  const float* bb1 = (const float*)d_in[8];
  const float* W2  = (const float*)d_in[9];
  const float* bb2 = (const float*)d_in[10];
  const float* W3  = (const float*)d_in[11];
  const float* bb3 = (const float*)d_in[12];

  char* ws = (char*)d_ws;
  unsigned short* c1bf = (unsigned short*)(ws);                 //  52,428,800 B
  unsigned int*   wk2  = (unsigned int*)  (ws +  52428800);     //  10,616,832 B
  float* c2    = (float*)(ws +  63045632);                      //   9,437,184 B
  float* caps1 = (float*)(ws +  72482816);                      //   9,437,184 B
  float* uhat  = (float*)(ws +  81920000);                      // 188,743,680 B
  float* v1    = (float*)(ws + 270663680);
  float* s2    = (float*)(ws + 270827520);
  float* decin = (float*)(ws + 270991360);
  float* h1    = (float*)(ws + 271155200);
  float* h2    = (float*)(ws + 271679488);

  float* caps2_out = (float*)d_out;            // B*160
  float* ypred     = caps2_out + BATCH * 160;  // B
  float* dec_out   = ypred + BATCH;            // B*784

  conv1_kernel     <<<BATCH * 400, 256, 0, stream>>>(X, K1, c1b, c1bf);
  conv2w_kernel    <<<10368, 256, 0, stream>>>(K2, wk2);
  conv2_wmma_kernel<<<576, 256, 0, stream>>>(c1bf, wk2, c2b, c2);
  squash1_kernel   <<<1152, 256, 0, stream>>>(c2, caps1);
  uhat_wmma_kernel <<<dim3(1152, 16), 320, 0, stream>>>(caps1, Wc, uhat);
  v1_kernel        <<<BATCH, 160, 0, stream>>>(uhat, v1);
  routing_kernel   <<<BATCH, 256, 0, stream>>>(uhat, v1, s2);
  v2_kernel        <<<BATCH, 160, 0, stream>>>(s2, Y, caps2_out, ypred, decin);
  gemm_f32_wmma_kernel<<<dim3( 8, 16), 128, 0, stream>>>(decin, W1, bb1, h1,     256,  160,  512, 0);
  gemm_f32_wmma_kernel<<<dim3(16, 16), 128, 0, stream>>>(h1,    W2, bb2, h2,     256,  512, 1024, 0);
  gemm_f32_wmma_kernel<<<dim3(13, 16), 128, 0, stream>>>(h2,    W3, bb3, dec_out,256, 1024,  784, 1);
}
